// Attention_77781857731102
// MI455X (gfx1250) — compile-verified, matrix operand order checked by reading, not run on hardware
//
#include <hip/hip_runtime.h>

// ---------------------------------------------------------------------------
// Causal attention for MI455X (gfx1250, wave32, WMMA).
//   pass 0: convert activations f32->bf16, weights f32->bf16 transposed
//   pass 1: projections as register-blocked WMMA GEMMs (bf16 in, f32 accum)
//   pass 2: fused logits+softmax+PV per 16-query block, 225KB LDS,
//           Q tile staged via GLOBAL_LOAD_ASYNC_TO_LDS_B128 (asynccnt).
// ---------------------------------------------------------------------------

typedef __attribute__((ext_vector_type(4)))  float          f32x4;
typedef __attribute__((ext_vector_type(8)))  float          f32x8;
typedef __attribute__((ext_vector_type(4)))  int            i32x4;
typedef __attribute__((ext_vector_type(4)))  unsigned short us4;
typedef __attribute__((ext_vector_type(8)))  unsigned short us8;
typedef __attribute__((ext_vector_type(16))) unsigned short us16;
typedef __attribute__((ext_vector_type(16))) __bf16         bf16x16;

constexpr int BB = 4;
constexpr int SS = 2048;
constexpr int DD = 1024;
constexpr int BS = BB * SS;  // 8192 tokens

// float -> bf16 (RNE)
__device__ inline unsigned short f2bf(float f) {
  __bf16 h = (__bf16)f;
  return __builtin_bit_cast(unsigned short, h);
}

__device__ inline bf16x16 pack16(us8 lo, us8 hi) {
  us16 c = __builtin_shufflevector(lo, hi, 0,1,2,3,4,5,6,7,8,9,10,11,12,13,14,15);
  return __builtin_bit_cast(bf16x16, c);
}

// Load one 16x32 bf16 WMMA operand (A-layout; lane l15 = row, hi = K group):
// two contiguous 16B chunks at +0 and +16 ushorts.
__device__ inline bf16x16 ld_op(const unsigned short* p) {
  return pack16(*(const us8*)p, *(const us8*)(p + 16));
}

__device__ inline f32x8 wmma_bf16(bf16x16 a, bf16x16 b, f32x8 c) {
  // (neg_a, A, neg_b, B, c_mod, C, reuse_a, reuse_b)
  return __builtin_amdgcn_wmma_f32_16x16x32_bf16(false, a, false, b, (short)0, c,
                                                 false, false);
}

// ---------------------------------------------------------------------------
// Pass 0a: elementwise f32 -> bf16 (8 elems/thread, b128 in / b128 out)
// ---------------------------------------------------------------------------
__global__ __launch_bounds__(256) void cvt_f32_bf16(
    const float* __restrict__ x, unsigned short* __restrict__ y) {
  size_t i = ((size_t)blockIdx.x * 256 + threadIdx.x) * 8;
  f32x4 a = *(const f32x4*)(x + i);
  f32x4 b = *(const f32x4*)(x + i + 4);
  us8 r;
  r[0]=f2bf(a[0]); r[1]=f2bf(a[1]); r[2]=f2bf(a[2]); r[3]=f2bf(a[3]);
  r[4]=f2bf(b[0]); r[5]=f2bf(b[1]); r[6]=f2bf(b[2]); r[7]=f2bf(b[3]);
  *(us8*)(y + i) = r;
}

// ---------------------------------------------------------------------------
// Pass 0b: W [k][n] f32 -> WT [n][k] bf16, 32x32 tiles via LDS transpose
// ---------------------------------------------------------------------------
__global__ __launch_bounds__(256) void cvt_w_transpose(
    const float* __restrict__ W, unsigned short* __restrict__ WT) {
  __shared__ unsigned short Ts[32 * 32];
  const int kb = blockIdx.x * 32, nb = blockIdx.y * 32;
  const int t = threadIdx.x;
  {
    const int k = t >> 3, n0 = (t & 7) * 4;
    f32x4 wv = *(const f32x4*)(W + (size_t)(kb + k) * DD + nb + n0);
    Ts[(n0 + 0) * 32 + k] = f2bf(wv[0]);
    Ts[(n0 + 1) * 32 + k] = f2bf(wv[1]);
    Ts[(n0 + 2) * 32 + k] = f2bf(wv[2]);
    Ts[(n0 + 3) * 32 + k] = f2bf(wv[3]);
  }
  __syncthreads();
  const int n = t >> 3, c0 = (t & 7) * 4;
  us4 o = *(const us4*)&Ts[n * 32 + c0];
  *(us4*)(WT + (size_t)(nb + n) * DD + kb + c0) = o;
}

// ---------------------------------------------------------------------------
// Pass 1: out = bf16(Xb @ WT^T + b).  Xb: [BS][DD] bf16, WT: [n][k] bf16.
// Block = 8 waves tiling 128 rows x 64 cols; each wave computes a 32x32
// macro-tile as 2x2 WMMA tiles (4 independent accumulator chains, operands
// reused across the 2x2 -> 2 b128 loads per v_wmma).
// transposed==0: out[token][col]; ==1: out[col][token] (V^T for PV GEMM).
// ---------------------------------------------------------------------------
__global__ __launch_bounds__(256) void proj_wmma(
    const unsigned short* __restrict__ Xb, const unsigned short* __restrict__ WT,
    const float* __restrict__ bias, unsigned short* __restrict__ out,
    int transposed) {
  const int tid  = threadIdx.x;
  const int wave = tid >> 5;
  const int lane = tid & 31;
  const int hi   = lane >> 4;
  const int l15  = lane & 15;

  const int rowBlk = blockIdx.x * 128 + (wave & 3) * 32;
  const int colBlk = blockIdx.y * 64 + (wave >> 2) * 32;

  const unsigned short* arow0 = Xb + (size_t)(rowBlk + l15) * DD + hi * 8;
  const unsigned short* arow1 = arow0 + (size_t)16 * DD;
  const unsigned short* brow0 = WT + (size_t)(colBlk + l15) * DD + hi * 8;
  const unsigned short* brow1 = brow0 + (size_t)16 * DD;

  f32x8 acc00 = {}, acc01 = {}, acc10 = {}, acc11 = {};
  for (int d0 = 0; d0 < DD; d0 += 32) {
    bf16x16 a0 = ld_op(arow0 + d0);
    bf16x16 a1 = ld_op(arow1 + d0);
    bf16x16 b0 = ld_op(brow0 + d0);
    bf16x16 b1 = ld_op(brow1 + d0);
    acc00 = wmma_bf16(a0, b0, acc00);
    acc01 = wmma_bf16(a0, b1, acc01);
    acc10 = wmma_bf16(a1, b0, acc10);
    acc11 = wmma_bf16(a1, b1, acc11);
  }

  const int col0 = colBlk + l15;
  const float bb0 = bias[col0];
  const float bb1 = bias[col0 + 16];
  const int tok0 = rowBlk + hi * 8;  // C/D: M = r + 8*hi
  for (int r = 0; r < 8; ++r) {
    const int ta = tok0 + r, tb = ta + 16;
    unsigned short q00 = f2bf(acc00[r] + bb0);
    unsigned short q01 = f2bf(acc01[r] + bb1);
    unsigned short q10 = f2bf(acc10[r] + bb0);
    unsigned short q11 = f2bf(acc11[r] + bb1);
    if (transposed) {
      out[(size_t)col0 * BS + ta]        = q00;
      out[(size_t)(col0 + 16) * BS + ta] = q01;
      out[(size_t)col0 * BS + tb]        = q10;
      out[(size_t)(col0 + 16) * BS + tb] = q11;
    } else {
      out[(size_t)ta * DD + col0]      = q00;
      out[(size_t)ta * DD + col0 + 16] = q01;
      out[(size_t)tb * DD + col0]      = q10;
      out[(size_t)tb * DD + col0 + 16] = q11;
    }
  }
}

// ---------------------------------------------------------------------------
// Pass 2: per (batch, 16-query block): logits -> softmax -> O = P @ V.
// LDS: Slog 16x2048 f32 | Sp 16x2048 bf16 | Qs 16x1024 bf16 | Sred 16x16 f32
// ---------------------------------------------------------------------------
#define SMEM_P    (16 * 2048 * 4)              // 131072
#define SMEM_Q    (SMEM_P + 16 * 2048 * 2)     // 196608
#define SMEM_RED  (SMEM_Q + 16 * 1024 * 2)     // 229376
#define SMEM_TOT  (SMEM_RED + 16 * 16 * 4)     // 230400 bytes (<320KB/WGP)

__global__ __launch_bounds__(256) void attn_wmma(
    const unsigned short* __restrict__ qp, const unsigned short* __restrict__ kp,
    const unsigned short* __restrict__ vpT, float* __restrict__ outO,
    float* __restrict__ outA) {
  extern __shared__ char smem[];
  float*          Slog = (float*)(smem);
  unsigned short* Sp   = (unsigned short*)(smem + SMEM_P);
  unsigned short* Qs   = (unsigned short*)(smem + SMEM_Q);
  float*          Sred = (float*)(smem + SMEM_RED);

  const int tid  = threadIdx.x;
  const int wave = tid >> 5;
  const int lane = tid & 31;
  const int hi   = lane >> 4;
  const int l15  = lane & 15;

  const int b  = blockIdx.y;
  const int q0 = blockIdx.x * 16;

  // ---- Stage Q tile (16 x 1024 bf16 = 32KB) into LDS via async DMA ----
  for (int i = 0; i < 8; ++i) {
    const int c   = tid + i * 256;
    const int row = c >> 7;             // 128 x 16B chunks per row
    const int off = (c & 127) * 8;      // ushort offset within row
    const unsigned short* g = qp + (size_t)(b * SS + q0 + row) * DD + off;
    unsigned short* l = Qs + row * DD + off;
#if defined(__AMDGCN__) && __has_builtin(__builtin_amdgcn_global_load_async_to_lds_b128)
    __builtin_amdgcn_global_load_async_to_lds_b128(
        (__attribute__((address_space(1))) i32x4*)g,
        (__attribute__((address_space(3))) i32x4*)l, 0, 0);
#else
    *(us8*)l = *(const us8*)g;
#endif
  }
#if defined(__AMDGCN__) && __has_builtin(__builtin_amdgcn_global_load_async_to_lds_b128)
#if __has_builtin(__builtin_amdgcn_s_wait_asynccnt)
  __builtin_amdgcn_s_wait_asynccnt(0);
#else
  asm volatile("s_wait_asynccnt 0x0" ::: "memory");
#endif
#endif
  __syncthreads();

  // ---- Phase 1: logits = (Qp @ Kp^T)/sqrt(D), causal; key tiles in pairs ----
  const unsigned short* qrow = Qs + l15 * DD + hi * 8;  // A from LDS
  for (int ktp = wave; ktp < SS / 32; ktp += 8) {
    const int kb = ktp * 32;
    if (kb > q0 + 15) {  // both key tiles fully masked
      for (int r = 0; r < 8; ++r) {
        Slog[(r + 8 * hi) * SS + kb + l15]      = -1.0e9f;
        Slog[(r + 8 * hi) * SS + kb + 16 + l15] = -1.0e9f;
      }
      continue;
    }
    f32x8 acc0 = {}, acc1 = {};
    const unsigned short* krow0 =
        kp + ((size_t)(b * SS + kb + l15)) * DD + hi * 8;
    const unsigned short* krow1 = krow0 + (size_t)16 * DD;
    for (int d0 = 0; d0 < DD; d0 += 32) {
      bf16x16 a  = ld_op(qrow + d0);      // reused across both key tiles
      bf16x16 k0 = ld_op(krow0 + d0);
      bf16x16 k1 = ld_op(krow1 + d0);
      acc0 = wmma_bf16(a, k0, acc0);
      acc1 = wmma_bf16(a, k1, acc1);
    }
    for (int r = 0; r < 8; ++r) {
      const int m = r + 8 * hi;
      const int key0 = kb + l15, key1 = key0 + 16;
      float lg0 = acc0[r] * 0.03125f;      // 1/sqrt(1024)
      float lg1 = acc1[r] * 0.03125f;
      if (key0 > q0 + m) lg0 = -1.0e9f;    // causal mask (triu k=1)
      if (key1 > q0 + m) lg1 = -1.0e9f;
      Slog[m * SS + key0] = lg0;
      Slog[m * SS + key1] = lg1;
    }
  }
  __syncthreads();

  // ---- Phase 2: row softmax (vectorized), write attn f32 + P bf16 ----
  {
    const int row = tid >> 4;  // 16 rows x 16 threads, 128 keys each
    const int sl  = tid & 15;
    f32x4* lr = (f32x4*)(Slog + row * SS + sl * 128);
    float mx = -3.4e38f;
    for (int i = 0; i < 32; ++i) {
      f32x4 v = lr[i];
      mx = fmaxf(mx, fmaxf(fmaxf(v[0], v[1]), fmaxf(v[2], v[3])));
    }
    Sred[row * 16 + sl] = mx;
    __syncthreads();
    float m2 = -3.4e38f;
    for (int i = 0; i < 16; ++i) m2 = fmaxf(m2, Sred[row * 16 + i]);
    __syncthreads();
    float s = 0.f;
    for (int i = 0; i < 32; ++i) {
      f32x4 v = lr[i];
      v[0] = __expf(v[0] - m2); v[1] = __expf(v[1] - m2);
      v[2] = __expf(v[2] - m2); v[3] = __expf(v[3] - m2);
      lr[i] = v;
      s += (v[0] + v[1]) + (v[2] + v[3]);
    }
    Sred[row * 16 + sl] = s;
    __syncthreads();
    float tot = 0.f;
    for (int i = 0; i < 16; ++i) tot += Sred[row * 16 + i];
    const float inv = 1.0f / tot;  // diagonal always unmasked -> tot > 0
    f32x4* arow = (f32x4*)(outA + ((size_t)(b * SS + q0 + row)) * SS + sl * 128);
    us4*   prow = (us4*)(Sp + row * SS + sl * 128);
    for (int i = 0; i < 32; ++i) {
      f32x4 p = lr[i];
      p[0] *= inv; p[1] *= inv; p[2] *= inv; p[3] *= inv;
      arow[i] = p;
      us4 pb; pb[0]=f2bf(p[0]); pb[1]=f2bf(p[1]); pb[2]=f2bf(p[2]); pb[3]=f2bf(p[3]);
      prow[i] = pb;
    }
  }
  __syncthreads();

  // ---- Phase 3: O = P @ Vp (vpT[d][token]); d tiles in pairs ----
  const int nsteps = (q0 + 16 + 31) >> 5;  // keys beyond q0+15 have P == 0
  const unsigned short* prowA = Sp + l15 * SS + hi * 8;
  for (int i4 = 0; i4 < 4; ++i4) {
    const int dcol = (wave + i4 * 8) * 32;
    f32x8 acc0 = {}, acc1 = {};
    const unsigned short* vcol0 =
        vpT + (size_t)(dcol + l15) * BS + b * SS + hi * 8;
    const unsigned short* vcol1 = vcol0 + (size_t)16 * BS;
    for (int s = 0; s < nsteps; ++s) {
      const int k0 = s * 32;
      bf16x16 a  = ld_op(prowA + k0);     // reused across both d tiles
      bf16x16 v0 = ld_op(vcol0 + k0);
      bf16x16 v1 = ld_op(vcol1 + k0);
      acc0 = wmma_bf16(a, v0, acc0);
      acc1 = wmma_bf16(a, v1, acc1);
    }
    float* orow = outO + ((size_t)(b * SS + q0 + 8 * hi)) * DD + dcol + l15;
    for (int r = 0; r < 8; ++r) {
      orow[(size_t)r * DD]      = acc0[r];
      orow[(size_t)r * DD + 16] = acc1[r];
    }
  }
}

// ---------------------------------------------------------------------------
extern "C" void kernel_launch(void* const* d_in, const int* in_sizes, int n_in,
                              void* d_out, int out_size, void* d_ws,
                              size_t ws_size, hipStream_t stream) {
  const float* v_in = (const float*)d_in[0];
  const float* k_in = (const float*)d_in[1];
  const float* q_in = (const float*)d_in[2];
  // d_in[3] (mask) unused: causal triu(k=1) mask computed analytically.
  const float* Wq = (const float*)d_in[4];
  const float* bq = (const float*)d_in[5];
  const float* Wk = (const float*)d_in[6];
  const float* bk = (const float*)d_in[7];
  const float* Wv = (const float*)d_in[8];
  const float* bv = (const float*)d_in[9];

  // Workspace layout (bf16): xb/wt reused across the three projections
  // (launches are in-order on `stream`).
  unsigned short* xb  = (unsigned short*)d_ws;          // [BS][DD]
  unsigned short* wt  = xb + (size_t)BS * DD;           // [DD][DD] (W^T)
  unsigned short* qp  = wt + (size_t)DD * DD;           // [BS][DD]
  unsigned short* kp  = qp + (size_t)BS * DD;           // [BS][DD]
  unsigned short* vpT = kp + (size_t)BS * DD;           // [DD][BS]

  float* outO = (float*)d_out;                // [B,S,D]
  float* outA = outO + (size_t)BB * SS * DD;  // [B,S,S]

  const int cvtBlocks = (BS * DD) / (256 * 8);  // 4096
  dim3 gw(DD / 32, DD / 32);                    // 32x32
  dim3 gp(BS / 128, DD / 64);                   // 64x16

  const float* X[3]  = {q_in, k_in, v_in};
  const float* W[3]  = {Wq, Wk, Wv};
  const float* Bv[3] = {bq, bk, bv};
  unsigned short* O[3] = {qp, kp, vpT};
  for (int m = 0; m < 3; ++m) {
    cvt_f32_bf16<<<cvtBlocks, 256, 0, stream>>>(X[m], xb);
    cvt_w_transpose<<<gw, 256, 0, stream>>>(W[m], wt);
    proj_wmma<<<gp, 256, 0, stream>>>(xb, wt, Bv[m], O[m], m == 2 ? 1 : 0);
  }

  (void)hipFuncSetAttribute((const void*)attn_wmma,
                            hipFuncAttributeMaxDynamicSharedMemorySize,
                            SMEM_TOT);
  attn_wmma<<<dim3(SS / 16, BB), 256, SMEM_TOT, stream>>>(qp, kp, vpT, outO,
                                                          outA);
}
